// MyGAT_54322746360234
// MI455X (gfx1250) — matrix operand order repaired and checked by reading, not compiled, not run on hardware
//
#include <hip/hip_runtime.h>
#include <cstdint>
#include <cstddef>

#define NNODES 50000
#define NEDGES 1000000
#define DIN 128
#define DHID 128
#define DOUT 64
#define NEG_SLOPE 0.2f

typedef __bf16 v16bf __attribute__((ext_vector_type(16)));
typedef __bf16 v8bf  __attribute__((ext_vector_type(8)));
typedef float  v8f   __attribute__((ext_vector_type(8)));
typedef float  v4f   __attribute__((ext_vector_type(4)));

// ---------------------------------------------------------------- fills
__global__ void fill_f32(float* __restrict__ p, float v, int n) {
    int i = blockIdx.x * blockDim.x + threadIdx.x;
    if (i < n) p[i] = v;
}
__global__ void fill_u32(unsigned* __restrict__ p, unsigned v, int n) {
    int i = blockIdx.x * blockDim.x + threadIdx.x;
    if (i < n) p[i] = v;
}

// ------------------------------------------------- order-preserving float<->uint
// monotone map so unsigned atomic-max == float max (handles negatives)
__device__ __forceinline__ unsigned fenc(float x) {
    unsigned u = __float_as_uint(x);
    return (u & 0x80000000u) ? ~u : (u | 0x80000000u);
}
__device__ __forceinline__ float fdec(unsigned e) {
    unsigned u = (e & 0x80000000u) ? (e ^ 0x80000000u) : ~e;
    return __uint_as_float(u);
}
#define ENC_NEG_INF 0x007FFFFFu   // fenc(-inf)

// ---------------------------------------------------------------- GEMM (WMMA bf16)
// H[M,N] = X[M,K] @ W[K,N]; fp32 in/out, bf16 v_wmma_f32_16x16x32_bf16, fp32 accum.
// K, N compile-time; M must be a multiple of 16 (50000 = 3125*16).
// Block = 256 threads = 8 waves; one wave per 16x16 output tile; all 8 waves share
// one 16-column W panel staged once into LDS (bf16, column-major).
template <int K, int N>
__global__ void gemm_wmma_bf16(const float* __restrict__ X,
                               const float* __restrict__ W,
                               float* __restrict__ H,
                               int M) {
    __shared__ __bf16 Bs[16 * K];          // [col][k], 16*K*2 bytes (4 KB for K=128)

    const int col0 = blockIdx.x * 16;

    // cooperative stage + fp32->bf16 convert of the W panel (once per block)
    for (int idx = threadIdx.x; idx < 16 * K; idx += blockDim.x) {
        const int c = idx / K;             // 0..15
        const int k = idx - c * K;         // 0..K-1
        Bs[c * K + k] = (__bf16)W[(size_t)k * N + col0 + c];
    }
    __syncthreads();

    const int lane = threadIdx.x & 31;
    const int wave = threadIdx.x >> 5;
    const int row0 = (blockIdx.y * 8 + wave) * 16;
    if (row0 >= M) return;                 // wave-uniform (EXEC stays full for WMMA)

    const int mrow = lane & 15;            // A row within tile / B-D col within tile
    const int hi   = lane >> 4;            // half-wave select
    const int aklo = hi * 8;               // A frag: K = aklo+0..7, aklo+16..23
    const int bklo = hi * 8;               // B frag: K = bklo+0..7, bklo+16..23

    const float*  Abase = X + (size_t)(row0 + mrow) * K;
    const __bf16* Bbase = &Bs[mrow * K];

    v8f acc = {};
    #pragma unroll
    for (int k = 0; k < K; k += 32) {
        // A fragment: 16 bf16 from row (row0+mrow), K chunks [k+aklo, +8) and [k+aklo+16, +8)
        const v4f a0 = *(const v4f*)(Abase + k + aklo);
        const v4f a1 = *(const v4f*)(Abase + k + aklo + 4);
        const v4f a2 = *(const v4f*)(Abase + k + aklo + 16);
        const v4f a3 = *(const v4f*)(Abase + k + aklo + 20);
        v16bf a;
        #pragma unroll
        for (int i = 0; i < 4; ++i) {
            a[i]      = (__bf16)a0[i];
            a[4 + i]  = (__bf16)a1[i];
            a[8 + i]  = (__bf16)a2[i];
            a[12 + i] = (__bf16)a3[i];
        }
        // B fragment: 2x 8 bf16 from LDS, K chunks [k+bklo, +8) and [k+bklo+16, +8) (16B-aligned) -> 2x ds_load_b128
        const v8bf blo = *(const v8bf*)(Bbase + k + bklo);
        const v8bf bhi = *(const v8bf*)(Bbase + k + bklo + 16);
        const v16bf b = __builtin_shufflevector(blo, bhi,
            0, 1, 2, 3, 4, 5, 6, 7, 8, 9, 10, 11, 12, 13, 14, 15);
        acc = __builtin_amdgcn_wmma_f32_16x16x32_bf16(
                  false, a, false, b, (short)0, acc, false, false);
    }

    // D layout: VGPR i -> M=i (lanes 0-15) / M=i+8 (lanes 16-31), col = lane&15.
    // Tiles are exact (M % 16 == 0): unconditional stores, shift-based addressing.
    float* Hbase = H + (size_t)(row0 + hi * 8) * N + col0 + mrow;
    #pragma unroll
    for (int i = 0; i < 8; ++i)
        Hbase[(size_t)i * N] = acc[i];
}

// ---------------------------------------------------------------- edge pass 1
// logits[e] = sum_c att[c]*leaky_relu(h[src][c]+h[dst][c]); atomic max into menc[dst]
template <int C>
__global__ void edge_logits_kernel(const float* __restrict__ H,
                                   const int* __restrict__ src,
                                   const int* __restrict__ dst,
                                   const float* __restrict__ att,
                                   float* __restrict__ logits,
                                   unsigned* __restrict__ menc,
                                   int E) {
    const int lane = threadIdx.x & 31;
    const int e = blockIdx.x * (blockDim.x >> 5) + (threadIdx.x >> 5);
    if (e >= E) return;
    const int s = src[e], d = dst[e];
    const float* hs = H + (size_t)s * C;
    const float* hd = H + (size_t)d * C;
    float acc = 0.f;
    #pragma unroll
    for (int c = lane; c < C; c += 32) {
        float v = hs[c] + hd[c];
        v = (v > 0.f) ? v : v * NEG_SLOPE;
        acc += v * att[c];
    }
    #pragma unroll
    for (int off = 16; off > 0; off >>= 1) acc += __shfl_xor(acc, off, 32);
    if (lane == 0) {
        logits[e] = acc;
        atomicMax(&menc[d], fenc(acc));
    }
}

// ---------------------------------------------------------------- edge pass 2
// lw[e] := exp(logits[e] - max[dst]); denom[dst] += lw[e]
__global__ void edge_exp_kernel(float* __restrict__ lw,
                                const int* __restrict__ dst,
                                const unsigned* __restrict__ menc,
                                float* __restrict__ denom, int E) {
    int e = blockIdx.x * blockDim.x + threadIdx.x;
    if (e >= E) return;
    int d = dst[e];
    float ee = __expf(lw[e] - fdec(menc[d]));
    lw[e] = ee;
    atomicAdd(&denom[d], ee);
}

// ---------------------------------------------------------------- edge pass 3
// out[dst] += (lw[e]/denom[dst]) * h[src]
template <int C>
__global__ void edge_agg_kernel(const float* __restrict__ H,
                                const int* __restrict__ src,
                                const int* __restrict__ dst,
                                const float* __restrict__ ew,
                                const float* __restrict__ denom,
                                float* __restrict__ out, int E) {
    const int lane = threadIdx.x & 31;
    const int e = blockIdx.x * (blockDim.x >> 5) + (threadIdx.x >> 5);
    if (e >= E) return;
    const int s = src[e], d = dst[e];
    const float* hs = H + (size_t)s * C;
    // kick the h[src] row toward the WGP while the scalar ew/denom loads resolve
    __builtin_prefetch(hs + lane, 0, 3);            // -> global_prefetch_b8
    const float alpha = ew[e] / denom[d];
    float* od = out + (size_t)d * C;
    #pragma unroll
    for (int c = lane; c < C; c += 32)
        atomicAdd(&od[c], alpha * hs[c]);
}

// ---------------------------------------------------------------- epilogue
__global__ void bias_act_kernel(float* __restrict__ y, const float* __restrict__ b,
                                int total, int C, int do_relu) {
    int i = blockIdx.x * blockDim.x + threadIdx.x;
    if (i >= total) return;
    float v = y[i] + b[i & (C - 1)];   // C is a power of two (128 / 64)
    if (do_relu) v = (v > 0.f) ? v : 0.f;
    y[i] = v;
}

// ================================================================ host
static inline size_t alignUp(size_t x, size_t a) { return (x + a - 1) & ~(a - 1); }

extern "C" void kernel_launch(void* const* d_in, const int* in_sizes, int n_in,
                              void* d_out, int out_size, void* d_ws, size_t ws_size,
                              hipStream_t stream) {
    (void)in_sizes; (void)n_in; (void)out_size; (void)ws_size;

    const float*  x    = (const float*)d_in[0];
    const int*    ei   = (const int*)d_in[1];     // [2, E]
    const float*  W1   = (const float*)d_in[2];
    const float*  att1 = (const float*)d_in[3];
    const float*  b1   = (const float*)d_in[4];
    const float*  W2   = (const float*)d_in[5];
    const float*  att2 = (const float*)d_in[6];
    const float*  b2   = (const float*)d_in[7];
    const int*    src  = ei;
    const int*    dst  = ei + NEDGES;
    float*        out  = (float*)d_out;           // [NNODES, DOUT]

    // workspace layout
    char* ws = (char*)d_ws;
    size_t off = 0;
    float*    h1    = (float*)(ws + off); off = alignUp(off + (size_t)NNODES * DHID * 4, 256);
    float*    agg1  = (float*)(ws + off); off = alignUp(off + (size_t)NNODES * DHID * 4, 256);
    float*    h2    = (float*)(ws + off); off = alignUp(off + (size_t)NNODES * DOUT * 4, 256);
    float*    lw    = (float*)(ws + off); off = alignUp(off + (size_t)NEDGES * 4, 256);
    unsigned* menc  = (unsigned*)(ws + off); off = alignUp(off + (size_t)NNODES * 4, 256);
    float*    denom = (float*)(ws + off); off = alignUp(off + (size_t)NNODES * 4, 256);

    const int T = 256;
    const int rowTiles = (NNODES + 15) / 16;            // 3125 (exact tiles)
    const dim3 gemmBlk(T);
    const dim3 gemmGrid1(DHID / 16, (rowTiles + 7) / 8);
    const dim3 gemmGrid2(DOUT / 16, (rowTiles + 7) / 8);
    const int edgeWaveBlocks = (NEDGES + 7) / 8;        // 8 waves/block, 1 edge/wave
    const int edgeThrBlocks  = (NEDGES + T - 1) / T;

    // ---------------- layer 1 ----------------
    gemm_wmma_bf16<DIN, DHID><<<gemmGrid1, gemmBlk, 0, stream>>>(x, W1, h1, NNODES);

    fill_f32<<<((NNODES * DHID) + T - 1) / T, T, 0, stream>>>(agg1, 0.f, NNODES * DHID);
    fill_u32<<<(NNODES + T - 1) / T, T, 0, stream>>>(menc, ENC_NEG_INF, NNODES);
    fill_f32<<<(NNODES + T - 1) / T, T, 0, stream>>>(denom, 0.f, NNODES);

    edge_logits_kernel<DHID><<<edgeWaveBlocks, T, 0, stream>>>(h1, src, dst, att1, lw, menc, NEDGES);
    edge_exp_kernel<<<edgeThrBlocks, T, 0, stream>>>(lw, dst, menc, denom, NEDGES);
    edge_agg_kernel<DHID><<<edgeWaveBlocks, T, 0, stream>>>(h1, src, dst, lw, denom, agg1, NEDGES);
    bias_act_kernel<<<((NNODES * DHID) + T - 1) / T, T, 0, stream>>>(agg1, b1, NNODES * DHID, DHID, 1);

    // ---------------- layer 2 ----------------
    gemm_wmma_bf16<DHID, DOUT><<<gemmGrid2, gemmBlk, 0, stream>>>(agg1, W2, h2, NNODES);

    fill_f32<<<((NNODES * DOUT) + T - 1) / T, T, 0, stream>>>(out, 0.f, NNODES * DOUT);
    fill_u32<<<(NNODES + T - 1) / T, T, 0, stream>>>(menc, ENC_NEG_INF, NNODES);
    fill_f32<<<(NNODES + T - 1) / T, T, 0, stream>>>(denom, 0.f, NNODES);

    edge_logits_kernel<DOUT><<<edgeWaveBlocks, T, 0, stream>>>(h2, src, dst, att2, lw, menc, NEDGES);
    edge_exp_kernel<<<edgeThrBlocks, T, 0, stream>>>(lw, dst, menc, denom, NEDGES);
    edge_agg_kernel<DOUT><<<edgeWaveBlocks, T, 0, stream>>>(h2, src, dst, lw, denom, out, NEDGES);
    bias_act_kernel<<<((NNODES * DOUT) + T - 1) / T, T, 0, stream>>>(out, b2, NNODES * DOUT, DOUT, 0);
}
